// MoEMLP_32504312496611
// MI455X (gfx1250) — compile-verified, matrix operand order checked by reading, not run on hardware
//
#include <hip/hip_runtime.h>
#include <hip/hip_bf16.h>
#include <math.h>

// Problem constants (from reference): B=2, T=2048, H=1024, E=8, F=4096, top-2.
#define NTOK   4096          // B*T tokens
#define HID    1024
#define NEXP   8
#define FFN    4096
#define TOPK   2
#define RCAP   8704          // >= NTOK*TOPK + NEXP*63, multiple of 64
#define BM     64            // block tile M
#define BN     128           // block tile N
#define BK     32            // K step (bf16 wmma K)
#define MTILES (RCAP / BM)   // 136 row-tiles of 64

#define LDS_ROW   80                    // (BK + 8 pad) * 2 bytes -> conflict-free b128 reads
#define A_BYTES   (BM * LDS_ROW)        // 5120
#define B_BYTES   (BN * LDS_ROW)        // 10240
#define BUF_BYTES (A_BYTES + B_BYTES)   // 15360 per buffer, x2 = 30 KB LDS

typedef __attribute__((ext_vector_type(16))) __bf16 v16bf;
typedef __attribute__((ext_vector_type(8)))  __bf16 v8bf;
typedef __attribute__((ext_vector_type(4)))  __bf16 v4bf;
typedef __attribute__((ext_vector_type(8)))  float  v8f;
typedef __attribute__((ext_vector_type(4)))  float  v4f;
typedef __attribute__((ext_vector_type(4)))  int    v4i;

typedef v4i __attribute__((address_space(1)))* gas_v4i;   // global (AS1) int4*
typedef v4i __attribute__((address_space(3)))* las_v4i;   // LDS    (AS3) int4*

union Frag16 { v16bf v; v8bf h[2]; };

// -------------------------------------------------- async Global->LDS plumbing

#if __has_builtin(__builtin_amdgcn_global_load_async_to_lds_b128)
#define HAS_ASYNC_LDS 1
#else
#define HAS_ASYNC_LDS 0
#endif

__device__ __forceinline__ void cp_b128(char* lds_dst, const __bf16* g) {
#if HAS_ASYNC_LDS
    // Builtin wants (AS1 int4*, AS3 int4*, imm offset, imm cpol).
    // Build both pointers via inttoptr: global VA for AS1; low 32 bits of the
    // generic shared pointer are the wave-relative LDS byte offset for AS3.
    __builtin_amdgcn_global_load_async_to_lds_b128(
        (gas_v4i)(unsigned long long)g,
        (las_v4i)(unsigned long long)(unsigned)(unsigned long long)lds_dst,
        0, 0);
#else
    *(v8bf*)lds_dst = *(const v8bf*)g;   // synchronous fallback (still LDS-staged)
#endif
}

__device__ __forceinline__ void wait_async6() {
#if HAS_ASYNC_LDS
    asm volatile("s_wait_asynccnt 0x6" ::: "memory");
#endif
}

__device__ __forceinline__ void wait_async0() {
#if HAS_ASYNC_LDS
    asm volatile("s_wait_asynccnt 0x0" ::: "memory");
#endif
}

// Each of the 128 threads stages 6 x b128 per K-step:
//   A: row (tid>>1), two chunks selected by (tid&1)  (64 rows x 64B)
//   B: row tid, four chunks                          (128 rows x 64B)
__device__ __forceinline__ void stage_tile(const __bf16* ag, const __bf16* bg,
                                           char* al, char* bl, int k0) {
    cp_b128(al,      ag + k0);
    cp_b128(al + 16, ag + k0 + 8);
    #pragma unroll
    for (int c = 0; c < 4; ++c)
        cp_b128(bl + c * 16, bg + k0 + c * 8);
}

// ---------------------------------------------------------------- utilities

__global__ void zero_meta_kernel(int* __restrict__ counts, int* __restrict__ cursor,
                                 int* __restrict__ token_of) {
    int i = blockIdx.x * blockDim.x + threadIdx.x;
    if (i < NEXP) { counts[i] = 0; cursor[i] = 0; }
    for (int r = i; r < RCAP; r += gridDim.x * blockDim.x) token_of[r] = 0;
}

__global__ void f32_to_bf16_kernel(const float* __restrict__ src,
                                   __bf16* __restrict__ dst, int n4) {
    int i = blockIdx.x * blockDim.x + threadIdx.x;
    int stride = gridDim.x * blockDim.x;
    for (; i < n4; i += stride) {
        v4f x = ((const v4f*)src)[i];
        v4bf o;
        o[0] = (__bf16)x[0]; o[1] = (__bf16)x[1];
        o[2] = (__bf16)x[2]; o[3] = (__bf16)x[3];
        ((v4bf*)dst)[i] = o;
    }
}

// ---------------------------------------------------------------- router

__global__ __launch_bounds__(128)
void router_kernel(const float* __restrict__ x, const float* __restrict__ rw,
                   int* __restrict__ counts, int* __restrict__ top_idx,
                   float* __restrict__ gates) {
    const int lane = threadIdx.x & 31;
    const int wav  = threadIdx.x >> 5;
    const int t    = blockIdx.x * (blockDim.x >> 5) + wav;
    if (t >= NTOK) return;

    float acc[NEXP];
    #pragma unroll
    for (int e = 0; e < NEXP; ++e) acc[e] = 0.f;

    const float* xp = x + (size_t)t * HID;
    for (int h = lane; h < HID; h += 32) {
        float xv = xp[h];
        #pragma unroll
        for (int e = 0; e < NEXP; ++e) acc[e] += xv * rw[e * HID + h];
    }
    #pragma unroll
    for (int e = 0; e < NEXP; ++e)
        for (int off = 16; off > 0; off >>= 1)
            acc[e] += __shfl_xor(acc[e], off, 32);

    if (lane == 0) {
        int i0 = 0;
        #pragma unroll
        for (int e = 1; e < NEXP; ++e) if (acc[e] > acc[i0]) i0 = e;   // ties -> lowest idx
        int i1 = (i0 == 0) ? 1 : 0;
        #pragma unroll
        for (int e = 0; e < NEXP; ++e)
            if (e != i0 && acc[e] > acc[i1]) i1 = e;
        float v0 = acc[i0], v1 = acc[i1];
        float g1 = 1.f / (1.f + expf(v0 - v1));    // softmax over {v0, v1}
        float g0 = 1.f - g1;
        top_idx[t * 2]     = i0;  top_idx[t * 2 + 1] = i1;
        gates[t * 2]       = g0;  gates[t * 2 + 1]   = g1;
        atomicAdd(&counts[i0], 1);
        atomicAdd(&counts[i1], 1);
    }
}

__global__ void offsets_kernel(const int* __restrict__ counts, int* __restrict__ offsets) {
    if (threadIdx.x == 0 && blockIdx.x == 0) {
        int run = 0;
        for (int e = 0; e < NEXP; ++e) {
            offsets[e] = run;
            run += (counts[e] + BM - 1) & ~(BM - 1);   // 64-aligned expert segments
        }
        offsets[NEXP] = run;
    }
}

__global__ void build_kernel(const int* __restrict__ top_idx, const int* __restrict__ offsets,
                             int* __restrict__ cursor, int* __restrict__ token_of,
                             int* __restrict__ slot_map) {
    int i = blockIdx.x * blockDim.x + threadIdx.x;
    if (i >= NTOK * TOPK) return;
    int e   = top_idx[i];
    int pos = atomicAdd(&cursor[e], 1);
    int r   = offsets[e] + pos;
    token_of[r] = i >> 1;
    slot_map[i] = r;
}

// ---------------------------------------------------------------- grouped WMMA GEMMs

__device__ __forceinline__ float gelu_exact(float v) {
    return 0.5f * v * (1.0f + erff(v * 0.70710678118654752f));
}

// h1[r, f] = gelu( sum_h xb[token_of[r], h] * w1b[e, f, h] + b1[e, f] )
__global__ __launch_bounds__(128)
void moe_ffn1_wmma_kernel(const __bf16* __restrict__ xb,      // [NTOK, HID]
                          const __bf16* __restrict__ w1b,     // [E, FFN, HID]
                          const float*  __restrict__ bias1,   // [E, FFN]
                          const int*    __restrict__ offsets, // [E+1]
                          const int*    __restrict__ token_of,// [RCAP]
                          __bf16*       __restrict__ h1b)     // [RCAP, FFN]
{
    __shared__ char smem[2 * BUF_BYTES];

    const int m0 = blockIdx.x * BM;
    if (m0 >= offsets[NEXP]) return;
    int e = 0;
    while (e < NEXP - 1 && m0 >= offsets[e + 1]) ++e;

    const int tid  = threadIdx.x;
    const int lane = tid & 31;
    const int wav  = tid >> 5;
    const int wi   = wav >> 1;            // wave M half (0..1)
    const int wj   = wav & 1;             // wave N half (0..1)
    const int lrow = lane & 15;
    const int kb2  = ((lane >> 4) * 8) * 2;   // per-lane K sub-offset in bytes
    const int nblk = blockIdx.y * BN;

    // copy-role pointers (gather token rows for A; pads read token 0)
    const __bf16* ag = xb + (size_t)token_of[m0 + (tid >> 1)] * HID + (tid & 1) * 16;
    const __bf16* bg = w1b + ((size_t)e * FFN + nblk + tid) * HID;
    char* al = smem + (tid >> 1) * LDS_ROW + (tid & 1) * 32;
    char* bl = smem + A_BYTES + tid * LDS_ROW;

    const v8f zf = {0.f, 0.f, 0.f, 0.f, 0.f, 0.f, 0.f, 0.f};
    v8f acc[2][4];
    #pragma unroll
    for (int i = 0; i < 2; ++i)
        #pragma unroll
        for (int j = 0; j < 4; ++j) acc[i][j] = zf;

    stage_tile(ag, bg, al, bl, 0);        // prologue: fill buffer 0

    const int KT = HID / BK;
    for (int kt = 0; kt < KT; ++kt) {
        const int cur = kt & 1;
        if (kt + 1 < KT) {
            const int nxt = (kt + 1) & 1;
            stage_tile(ag, bg, al + nxt * BUF_BYTES, bl + nxt * BUF_BYTES, (kt + 1) * BK);
            wait_async6();                // our 6 copies for `cur` are done
        } else {
            wait_async0();
        }
        __syncthreads();

        const char* sa = smem + cur * BUF_BYTES;
        const char* sb = sa + A_BYTES;
        Frag16 fa[2], fb[4];
        #pragma unroll
        for (int i = 0; i < 2; ++i) {
            const char* p = sa + (wi * 32 + i * 16 + lrow) * LDS_ROW + kb2;
            fa[i].h[0] = *(const v8bf*)p;
            fa[i].h[1] = *(const v8bf*)(p + 32);
        }
        #pragma unroll
        for (int j = 0; j < 4; ++j) {
            const char* p = sb + (wj * 64 + j * 16 + lrow) * LDS_ROW + kb2;
            fb[j].h[0] = *(const v8bf*)p;
            fb[j].h[1] = *(const v8bf*)(p + 32);
        }
        #pragma unroll
        for (int j = 0; j < 4; ++j) {
            acc[0][j] = __builtin_amdgcn_wmma_f32_16x16x32_bf16(
                false, fa[0].v, false, fb[j].v, (short)0, acc[0][j], false, false);
            acc[1][j] = __builtin_amdgcn_wmma_f32_16x16x32_bf16(
                false, fa[1].v, false, fb[j].v, (short)0, acc[1][j], false, false);
        }
        __syncthreads();
    }

    // C/D layout: VGPR v -> row m0 + wi*32 + i*16 + v + (lane>>4)*8, col nblk + wj*64 + j*16 + lrow
    #pragma unroll
    for (int i = 0; i < 2; ++i) {
        const int rbase = m0 + wi * 32 + i * 16 + (lane >> 4) * 8;
        #pragma unroll
        for (int j = 0; j < 4; ++j) {
            const int col = nblk + wj * 64 + j * 16 + lrow;
            const float bv = bias1[e * FFN + col];
            #pragma unroll
            for (int v = 0; v < 8; ++v) {
                float val = gelu_exact(acc[i][j][v] + bv);
                h1b[(size_t)(rbase + v) * FFN + col] = (__bf16)val;
            }
        }
    }
}

// ya[r, h] = sum_f h1b[r, f] * w2b[e, h, f] + b2[e, h]
__global__ __launch_bounds__(128)
void moe_ffn2_wmma_kernel(const __bf16* __restrict__ h1b,     // [RCAP, FFN]
                          const __bf16* __restrict__ w2b,     // [E, HID, FFN]
                          const float*  __restrict__ bias2,   // [E, HID]
                          const int*    __restrict__ offsets, // [E+1]
                          float*        __restrict__ ya)      // [RCAP, HID]
{
    __shared__ char smem[2 * BUF_BYTES];

    const int m0 = blockIdx.x * BM;
    if (m0 >= offsets[NEXP]) return;
    int e = 0;
    while (e < NEXP - 1 && m0 >= offsets[e + 1]) ++e;

    const int tid  = threadIdx.x;
    const int lane = tid & 31;
    const int wav  = tid >> 5;
    const int wi   = wav >> 1;
    const int wj   = wav & 1;
    const int lrow = lane & 15;
    const int kb2  = ((lane >> 4) * 8) * 2;
    const int nblk = blockIdx.y * BN;

    const __bf16* ag = h1b + (size_t)(m0 + (tid >> 1)) * FFN + (tid & 1) * 16;
    const __bf16* bg = w2b + ((size_t)e * HID + nblk + tid) * FFN;
    char* al = smem + (tid >> 1) * LDS_ROW + (tid & 1) * 32;
    char* bl = smem + A_BYTES + tid * LDS_ROW;

    const v8f zf = {0.f, 0.f, 0.f, 0.f, 0.f, 0.f, 0.f, 0.f};
    v8f acc[2][4];
    #pragma unroll
    for (int i = 0; i < 2; ++i)
        #pragma unroll
        for (int j = 0; j < 4; ++j) acc[i][j] = zf;

    stage_tile(ag, bg, al, bl, 0);

    const int KT = FFN / BK;
    for (int kt = 0; kt < KT; ++kt) {
        const int cur = kt & 1;
        if (kt + 1 < KT) {
            const int nxt = (kt + 1) & 1;
            stage_tile(ag, bg, al + nxt * BUF_BYTES, bl + nxt * BUF_BYTES, (kt + 1) * BK);
            wait_async6();
        } else {
            wait_async0();
        }
        __syncthreads();

        const char* sa = smem + cur * BUF_BYTES;
        const char* sb = sa + A_BYTES;
        Frag16 fa[2], fb[4];
        #pragma unroll
        for (int i = 0; i < 2; ++i) {
            const char* p = sa + (wi * 32 + i * 16 + lrow) * LDS_ROW + kb2;
            fa[i].h[0] = *(const v8bf*)p;
            fa[i].h[1] = *(const v8bf*)(p + 32);
        }
        #pragma unroll
        for (int j = 0; j < 4; ++j) {
            const char* p = sb + (wj * 64 + j * 16 + lrow) * LDS_ROW + kb2;
            fb[j].h[0] = *(const v8bf*)p;
            fb[j].h[1] = *(const v8bf*)(p + 32);
        }
        #pragma unroll
        for (int j = 0; j < 4; ++j) {
            acc[0][j] = __builtin_amdgcn_wmma_f32_16x16x32_bf16(
                false, fa[0].v, false, fb[j].v, (short)0, acc[0][j], false, false);
            acc[1][j] = __builtin_amdgcn_wmma_f32_16x16x32_bf16(
                false, fa[1].v, false, fb[j].v, (short)0, acc[1][j], false, false);
        }
        __syncthreads();
    }

    #pragma unroll
    for (int i = 0; i < 2; ++i) {
        const int rbase = m0 + wi * 32 + i * 16 + (lane >> 4) * 8;
        #pragma unroll
        for (int j = 0; j < 4; ++j) {
            const int col = nblk + wj * 64 + j * 16 + lrow;
            const float bv = bias2[e * HID + col];
            #pragma unroll
            for (int v = 0; v < 8; ++v)
                ya[(size_t)(rbase + v) * HID + col] = acc[i][j][v] + bv;
        }
    }
}

// ---------------------------------------------------------------- merge (deterministic)

__global__ void merge_kernel(const float* __restrict__ ya, const int* __restrict__ slot_map,
                             const float* __restrict__ gates, float* __restrict__ out) {
    const int W = HID / 4;
    int gid = blockIdx.x * blockDim.x + threadIdx.x;
    if (gid >= NTOK * W) return;
    int t = gid / W, c = gid % W;
    int r0 = slot_map[t * 2], r1 = slot_map[t * 2 + 1];
    float g0 = gates[t * 2], g1 = gates[t * 2 + 1];
    v4f y0 = ((const v4f*)(ya + (size_t)r0 * HID))[c];
    v4f y1 = ((const v4f*)(ya + (size_t)r1 * HID))[c];
    v4f o;
    #pragma unroll
    for (int k = 0; k < 4; ++k) o[k] = g0 * y0[k] + g1 * y1[k];
    ((v4f*)(out + (size_t)t * HID))[c] = o;
}

// ---------------------------------------------------------------- launch

extern "C" void kernel_launch(void* const* d_in, const int* in_sizes, int n_in,
                              void* d_out, int out_size, void* d_ws, size_t ws_size,
                              hipStream_t stream) {
    const float* x   = (const float*)d_in[0];  // [NTOK, HID]
    const float* rw  = (const float*)d_in[1];  // [E, HID]
    const float* w1  = (const float*)d_in[2];  // [E, FFN, HID]
    const float* b1  = (const float*)d_in[3];  // [E, FFN]
    const float* w2  = (const float*)d_in[4];  // [E, HID, FFN]
    const float* b2  = (const float*)d_in[5];  // [E, HID]
    float* out = (float*)d_out;

    // Workspace carve-up (~248 MB)
    char* ws = (char*)d_ws;
    size_t off = 0;
    auto carve = [&](size_t bytes) { char* p = ws + off; off = (off + bytes + 255) & ~(size_t)255; return p; };
    __bf16* xb   = (__bf16*)carve((size_t)NTOK * HID * 2);
    __bf16* w1b  = (__bf16*)carve((size_t)NEXP * FFN * HID * 2);
    __bf16* w2b  = (__bf16*)carve((size_t)NEXP * HID * FFN * 2);
    __bf16* h1b  = (__bf16*)carve((size_t)RCAP * FFN * 2);
    float*  ya   = (float*) carve((size_t)RCAP * HID * 4);
    int* counts   = (int*)carve(NEXP * 4);
    int* cursor   = (int*)carve(NEXP * 4);
    int* offsets  = (int*)carve((NEXP + 1) * 4);
    int* top_idx  = (int*)carve(NTOK * TOPK * 4);
    int* slot_map = (int*)carve(NTOK * TOPK * 4);
    int* token_of = (int*)carve(RCAP * 4);
    float* gates  = (float*)carve(NTOK * TOPK * 4);
    (void)ws_size; (void)in_sizes; (void)n_in; (void)out_size;

    zero_meta_kernel<<<34, 256, 0, stream>>>(counts, cursor, token_of);

    f32_to_bf16_kernel<<<2048, 256, 0, stream>>>(x,  xb,  NTOK * HID / 4);
    f32_to_bf16_kernel<<<8192, 256, 0, stream>>>(w1, w1b, NEXP * FFN * HID / 4);
    f32_to_bf16_kernel<<<8192, 256, 0, stream>>>(w2, w2b, NEXP * HID * FFN / 4);

    router_kernel<<<NTOK / 4, 128, 0, stream>>>(x, rw, counts, top_idx, gates);
    offsets_kernel<<<1, 32, 0, stream>>>(counts, offsets);
    build_kernel<<<(NTOK * TOPK + 255) / 256, 256, 0, stream>>>(top_idx, offsets, cursor,
                                                                token_of, slot_map);

    dim3 g1(MTILES, FFN / BN);
    moe_ffn1_wmma_kernel<<<g1, 128, 0, stream>>>(xb, w1b, b1, offsets, token_of, h1b);

    dim3 g2(MTILES, HID / BN);
    moe_ffn2_wmma_kernel<<<g2, 128, 0, stream>>>(h1b, w2b, b2, offsets, ya);

    merge_kernel<<<(NTOK * HID / 4 + 255) / 256, 256, 0, stream>>>(ya, slot_map, gates, out);
}
